// RAPMemory_31164282700405
// MI455X (gfx1250) — matrix operand order repaired and checked by reading, not compiled, not run on hardware
//
#include <hip/hip_runtime.h>

// ---------------- problem constants ----------------
#define Bq   16
#define Sq   128
#define Iq   576     // P + M
#define Uq   512     // 2*H
#define Hq   256
#define OUTq 64
#define MQ   (Bq*Sq) // 2048 rows for all GEMMs
#define UNF  6

typedef __attribute__((ext_vector_type(16))) _Float16 v16h;
typedef __attribute__((ext_vector_type(8)))  _Float16 v8h;
typedef __attribute__((ext_vector_type(8)))  float    v8f;

// ---------------- CDNA5 async global->LDS staging ----------------
// global_load_async_to_lds_b128: per-lane 16B, tracked on ASYNCcnt
// (CDNA5 ISA / CDNA5 ISA opcode 98).
__device__ __forceinline__ void async_load_b128(unsigned lds_off, const void* gptr) {
  unsigned long long ga = (unsigned long long)(uintptr_t)gptr;
  asm volatile("global_load_async_to_lds_b128 %0, %1, off"
               :: "v"(lds_off), "v"(ga) : "memory");
}
__device__ __forceinline__ void wait_async0() {
#if __has_builtin(__builtin_amdgcn_s_wait_asynccnt)
  __builtin_amdgcn_s_wait_asynccnt(0);
#else
  asm volatile("s_wait_asynccnt 0" ::: "memory");
#endif
}

// ---------------- WMMA helpers (A/B use contiguous k order: k = 16*half + s) ----
__device__ __forceinline__ v8f wmma_f16(v16h a, v16h b, v8f c) {
  return __builtin_amdgcn_wmma_f32_16x16x32_f16(
      /*neg_a=*/false, a, /*neg_b=*/false, b,
      /*c_mod=*/(short)0, c, /*reuse_a=*/false, /*reuse_b=*/false);
}

// A fragment 16x32 f16 at (row0,k0), row-major X, leading dim ld.
// Lane l: M = row0+(l&15); the 16 halves are contiguous at
// K = half*16 .. +16  ->  two 16B loads.
__device__ __forceinline__ v16h load_a(const _Float16* X, int ld, int row0, int k0) {
  const int lane = threadIdx.x & 31;
  const int half = lane >> 4;
  const _Float16* p = X + (size_t)(row0 + (lane & 15)) * ld + k0 + half * 16;
  v8h lo = *(const v8h*)p;          // K = half*16 .. +8
  v8h hi = *(const v8h*)(p + 8);    // K = half*16+8 .. +8
  return __builtin_shufflevector(lo, hi, 0,1,2,3,4,5,6,7,8,9,10,11,12,13,14,15);
}

// B fragment 32x16 where logical B[k][n] = T[n][k] (T row-major, ld).
// Lane l: N = col0+(l&15); 16 contiguous halves at K = k0 + half*16.
__device__ __forceinline__ v16h load_bt(const _Float16* T, int ld, int k0, int col0) {
  const int lane = threadIdx.x & 31;
  const _Float16* p = T + (size_t)(col0 + (lane & 15)) * ld + k0 + (lane >> 4) * 16;
  v8h lo = *(const v8h*)p;
  v8h hi = *(const v8h*)(p + 8);
  return __builtin_shufflevector(lo, hi, 0,1,2,3,4,5,6,7,8,9,10,11,12,13,14,15);
}

// B fragment 32x16 from row-major W (K x N), strided (used only for V tiles).
__device__ __forceinline__ v16h load_b(const _Float16* W, int ld, int k0, int col0) {
  const int lane = threadIdx.x & 31;
  const int col  = col0 + (lane & 15);
  const int half = lane >> 4;
  v16h b;
#pragma unroll
  for (int p = 0; p < 8; ++p) {
    int k = k0 + half * 16 + 2 * p;
    b[2 * p]     = W[k * ld + col];
    b[2 * p + 1] = W[(k + 1) * ld + col];
  }
  return b;
}

// ---------------- weight conversion (fp32 -> fp16, optionally transposed) ---
__global__ void cvt_f16_kernel(const float* __restrict__ src, _Float16* __restrict__ dst, int n) {
  int i = blockIdx.x * blockDim.x + threadIdx.x;
  if (i < n) dst[i] = (_Float16)src[i];
}
// src: R x C row-major  ->  dst: C x R row-major (transposed)
__global__ void cvt_f16_t_kernel(const float* __restrict__ src, _Float16* __restrict__ dst,
                                 int R, int C) {
  int i = blockIdx.x * blockDim.x + threadIdx.x;
  if (i < R * C) {
    int r = i / C, c = i % C;
    dst[(size_t)c * R + r] = (_Float16)src[i];
  }
}

// ---------------- LTC recurrence: one block per batch ----------------
__global__ __launch_bounds__(512)
void ltc_kernel(const float* __restrict__ x,
                const float* __restrict__ input_w, const float* __restrict__ input_b,
                const float* __restrict__ gleak,   const float* __restrict__ vleak,
                const float* __restrict__ cm,
                const float* __restrict__ sigma,   const float* __restrict__ mu,
                const float* __restrict__ w,       const float* __restrict__ erev,
                const float* __restrict__ mask,
                const float* __restrict__ s_sigma, const float* __restrict__ s_mu,
                const float* __restrict__ s_w,     const float* __restrict__ s_erev,
                const float* __restrict__ s_mask,
                const float* __restrict__ output_w, const float* __restrict__ output_b,
                float* __restrict__ ltc_f32, _Float16* __restrict__ ltc_f16,
                float* __restrict__ h_out) {
  __shared__ float inp[Iq];
  __shared__ float v[Uq];
  const int b = blockIdx.x;
  const int j = threadIdx.x;             // one output neuron per thread
  v[j] = 0.0f;
  const float cmt = cm[j] * (float)UNF;
  const float gl  = gleak[j];
  const float glvl = gl * vleak[j];
  __syncthreads();

  for (int t = 0; t < Sq; ++t) {
    for (int i = j; i < Iq; i += Uq)
      inp[i] = x[((size_t)b * Sq + t) * Iq + i] * input_w[i] + input_b[i];
    __syncthreads();

    float nums = 0.0f, dens = 0.0f;
    for (int i = 0; i < Iq; ++i) {
      const int idx = i * Uq + j;                       // coalesced across j
      float sg = 1.0f / (1.0f + __expf(-s_sigma[idx] * (inp[i] - s_mu[idx])));
      float a  = s_w[idx] * sg * s_mask[idx];
      nums += a * s_erev[idx];
      dens += a;
    }

#pragma unroll 1
    for (int u = 0; u < UNF; ++u) {
      float num = nums, den = dens;
      for (int i = 0; i < Uq; ++i) {
        const int idx = i * Uq + j;
        float sg = 1.0f / (1.0f + __expf(-sigma[idx] * (v[i] - mu[idx])));
        float a  = w[idx] * sg * mask[idx];
        num += a * erev[idx];
        den += a;
      }
      float vj = (cmt * v[j] + glvl + num) / (cmt + gl + den + 1e-8f);
      __syncthreads();
      v[j] = vj;
      __syncthreads();
    }

    if (j < Hq) {
      float o = v[j] * output_w[j] + output_b[j];
      size_t oidx = ((size_t)b * Sq + t) * Hq + j;
      ltc_f32[oidx] = o;
      ltc_f16[oidx] = (_Float16)o;
    }
    __syncthreads();
  }
  h_out[(size_t)b * Uq + j] = v[j];
}

// ---------------- WMMA GEMM with async LDS staging ----------------
// Y = act(X @ W + bias [+ residual]); X: MxK f16 row-major, WT: NxK f16
// (pre-transposed). Block = 4 waves, computes a 64x64 tile; wave wv does
// rows [64*by + 16*wv, +16) x all 64 cols. Tiles staged to LDS via
// global_load_async_to_lds_b128 each k-step.
#define GEMM_BK 32
template<bool RES, bool SILU, bool OUT32, bool OUT16>
__global__ __launch_bounds__(128)
void gemm_kernel(const _Float16* __restrict__ X, const _Float16* __restrict__ WT,
                 const float* __restrict__ bias, const float* __restrict__ residual,
                 float* __restrict__ Yf32, _Float16* __restrict__ Yf16,
                 int N, int K) {
  __shared__ _Float16 xtile[64 * GEMM_BK];   // 4 KB
  __shared__ _Float16 wtile[64 * GEMM_BK];   // 4 KB (transposed: [n][k])
  const int m0 = blockIdx.y * 64;
  const int n0 = blockIdx.x * 64;
  const int wv = threadIdx.y;                      // 0..3
  const int t  = wv * 32 + threadIdx.x;            // 0..127
  const unsigned xoff = (unsigned)(uintptr_t)xtile;
  const unsigned woff = (unsigned)(uintptr_t)wtile;

  v8f acc[4] = {{}, {}, {}, {}};
  for (int k0 = 0; k0 < K; k0 += GEMM_BK) {
    // stage: 64 rows x 32 halves = 256 x 16B chunks per array, 2 per thread
#pragma unroll
    for (int c = 0; c < 2; ++c) {
      int chunk = t + c * 128;
      int row = chunk >> 2, seg = chunk & 3;
      async_load_b128(xoff + (unsigned)(row * GEMM_BK + seg * 8) * 2,
                      X + (size_t)(m0 + row) * K + k0 + seg * 8);
      async_load_b128(woff + (unsigned)(row * GEMM_BK + seg * 8) * 2,
                      WT + (size_t)(n0 + row) * K + k0 + seg * 8);
    }
    wait_async0();
    __syncthreads();

    v16h a = load_a(xtile, GEMM_BK, wv * 16, 0);
#pragma unroll
    for (int tn = 0; tn < 4; ++tn) {
      v16h bfr = load_bt(wtile, GEMM_BK, 0, tn * 16);
      acc[tn] = wmma_f16(a, bfr, acc[tn]);
    }
    __syncthreads();
  }

  // epilogue: C layout — VGPR r holds (M = r + 8*(lane>>4), N = lane&15)
  const int lane = threadIdx.x & 31;
  const int mb   = (lane >> 4) * 8;
#pragma unroll
  for (int tn = 0; tn < 4; ++tn) {
    const int n = n0 + tn * 16 + (lane & 15);
    const float bn = bias[n];
#pragma unroll
    for (int r = 0; r < 8; ++r) {
      const int m = m0 + wv * 16 + mb + r;
      float val = acc[tn][r] + bn;
      if (RES)   val += residual[(size_t)m * N + n];
      if (SILU)  val = val * __builtin_amdgcn_rcpf(1.0f + __expf(-val));
      if (OUT32) Yf32[(size_t)m * N + n] = val;
      if (OUT16) Yf16[(size_t)m * N + n] = (_Float16)val;
    }
  }
}

// ---------------- attention: one block per (batch, head) --------------------
// 8 waves; wave w owns score rows [16w,16w+16): 8 register score tiles,
// in-wave shfl softmax, P spilled f16 to LDS, then P@V via WMMA from LDS.
__global__ __launch_bounds__(256)
void attn_kernel(const _Float16* __restrict__ Q, const _Float16* __restrict__ K,
                 const _Float16* __restrict__ V, _Float16* __restrict__ Mem) {
  __shared__ _Float16 P[Sq * Sq];        // 32 KB
  const int b  = blockIdx.x;
  const int hh = blockIdx.y;
  const int wv = threadIdx.y;
  const int lane = threadIdx.x & 31;
  const _Float16* Qp = Q + (size_t)b * Sq * Hq + hh * 64;
  const _Float16* Kp = K + (size_t)b * Sq * Hq + hh * 64;
  const _Float16* Vp = V + (size_t)b * Sq * Hq + hh * 64;
  const int row0 = wv * 16;

  // scores = (Q @ K^T) * rsqrt(DH), kept in registers
  v8f sc[8];
#pragma unroll
  for (int tj = 0; tj < 8; ++tj) {
    v8f acc = {};
#pragma unroll
    for (int k0 = 0; k0 < 64; k0 += 32) {
      v16h a  = load_a(Qp, Hq, row0, k0);
      v16h bb = load_bt(Kp, Hq, k0, tj * 16);
      acc = wmma_f16(a, bb, acc);
    }
    sc[tj] = acc * 0.125f;
  }

  // row softmax: row = row0 + vr + 8*(lane>>4); cols over (tj, lane&15);
  // reduce within the 16 lanes of the same half via shfl_xor (wave32).
  const int half = lane >> 4;
#pragma unroll
  for (int vr = 0; vr < 8; ++vr) {
    float m = -1e30f;
#pragma unroll
    for (int tj = 0; tj < 8; ++tj) m = fmaxf(m, sc[tj][vr]);
#pragma unroll
    for (int off = 1; off < 16; off <<= 1) m = fmaxf(m, __shfl_xor(m, off, 32));
    float e[8], s = 0.0f;
#pragma unroll
    for (int tj = 0; tj < 8; ++tj) { e[tj] = __expf(sc[tj][vr] - m); s += e[tj]; }
#pragma unroll
    for (int off = 1; off < 16; off <<= 1) s += __shfl_xor(s, off, 32);
    const float inv = __builtin_amdgcn_rcpf(s);
    const int row = row0 + vr + 8 * half;
    const int col = lane & 15;
#pragma unroll
    for (int tj = 0; tj < 8; ++tj)
      P[row * Sq + tj * 16 + col] = (_Float16)(e[tj] * inv);
  }
  __syncthreads();

  // out = P @ V_head : wave w computes rows [16w,16w+16), 4 col tiles
#pragma unroll
  for (int tn = 0; tn < 4; ++tn) {
    v8f acc = {};
#pragma unroll
    for (int k0 = 0; k0 < Sq; k0 += 32) {
      v16h a  = load_a(P, Sq, row0, k0);       // ds_load_b128 pairs
      v16h bb = load_b(Vp, Hq, k0, tn * 16);   // strided global
      acc = wmma_f16(a, bb, acc);
    }
    const int n  = hh * 64 + tn * 16 + (lane & 15);
    const int mb = (lane >> 4) * 8;
#pragma unroll
    for (int r = 0; r < 8; ++r)
      Mem[((size_t)b * Sq + row0 + mb + r) * Hq + n] = (_Float16)acc[r];
  }
}

// ---------------- launcher ----------------
extern "C" void kernel_launch(void* const* d_in, const int* in_sizes, int n_in,
                              void* d_out, int out_size, void* d_ws, size_t ws_size,
                              hipStream_t stream) {
  const float* x        = (const float*)d_in[0];
  const float* input_w  = (const float*)d_in[1];
  const float* input_b  = (const float*)d_in[2];
  const float* gleak    = (const float*)d_in[3];
  const float* vleak    = (const float*)d_in[4];
  const float* cm       = (const float*)d_in[5];
  const float* sigma    = (const float*)d_in[6];
  const float* mu       = (const float*)d_in[7];
  const float* w        = (const float*)d_in[8];
  const float* erev     = (const float*)d_in[9];
  const float* mask     = (const float*)d_in[10];
  const float* s_sigma  = (const float*)d_in[11];
  const float* s_mu     = (const float*)d_in[12];
  const float* s_w      = (const float*)d_in[13];
  const float* s_erev   = (const float*)d_in[14];
  const float* s_mask   = (const float*)d_in[15];
  const float* output_w = (const float*)d_in[16];
  const float* output_b = (const float*)d_in[17];
  const float* Wq = (const float*)d_in[18]; const float* bq = (const float*)d_in[19];
  const float* Wk = (const float*)d_in[20]; const float* bk = (const float*)d_in[21];
  const float* Wv = (const float*)d_in[22]; const float* bv = (const float*)d_in[23];
  const float* Wo = (const float*)d_in[24]; const float* bo = (const float*)d_in[25];
  const float* W1 = (const float*)d_in[26]; const float* b1 = (const float*)d_in[27];
  const float* W2 = (const float*)d_in[28]; const float* b2 = (const float*)d_in[29];

  float* out_f    = (float*)d_out;
  float* combined = out_f;                       // (B,S,H)  = 524288
  float* belief   = out_f + (size_t)MQ * Hq;     // (B,S,64) = 131072
  float* h_final  = belief + (size_t)MQ * OUTq;  // (B,U)    = 8192

  char* ws = (char*)d_ws; size_t off = 0;
  auto carve = [&](size_t bytes) { void* p = ws + off; off += (bytes + 255) & ~(size_t)255; return p; };
  float*    ltc_f32 = (float*)   carve((size_t)MQ * Hq * 4);
  _Float16* ltc_f16 = (_Float16*)carve((size_t)MQ * Hq * 2);
  _Float16* WqT = (_Float16*)carve((size_t)Hq * Hq * 2);     // N x K transposed
  _Float16* WkT = (_Float16*)carve((size_t)Hq * Hq * 2);
  _Float16* WvT = (_Float16*)carve((size_t)Hq * Hq * 2);
  _Float16* WoT = (_Float16*)carve((size_t)Hq * Hq * 2);
  _Float16* W1T = (_Float16*)carve((size_t)Hq * Hq * 2);
  _Float16* W2T = (_Float16*)carve((size_t)Hq * OUTq * 2);   // 64 x 256
  _Float16* Q16   = (_Float16*)carve((size_t)MQ * Hq * 2);
  _Float16* K16   = (_Float16*)carve((size_t)MQ * Hq * 2);
  _Float16* V16   = (_Float16*)carve((size_t)MQ * Hq * 2);
  _Float16* mem16 = (_Float16*)carve((size_t)MQ * Hq * 2);
  _Float16* cmb16 = (_Float16*)carve((size_t)MQ * Hq * 2);
  _Float16* x116  = (_Float16*)carve((size_t)MQ * Hq * 2);

  // weights -> fp16, transposed to N x K for contiguous B fragments
  const int nw = Hq * Hq;
  cvt_f16_t_kernel<<<(nw + 255) / 256, 256, 0, stream>>>(Wq, WqT, Hq, Hq);
  cvt_f16_t_kernel<<<(nw + 255) / 256, 256, 0, stream>>>(Wk, WkT, Hq, Hq);
  cvt_f16_t_kernel<<<(nw + 255) / 256, 256, 0, stream>>>(Wv, WvT, Hq, Hq);
  cvt_f16_t_kernel<<<(nw + 255) / 256, 256, 0, stream>>>(Wo, WoT, Hq, Hq);
  cvt_f16_t_kernel<<<(nw + 255) / 256, 256, 0, stream>>>(W1, W1T, Hq, Hq);
  cvt_f16_t_kernel<<<(Hq * OUTq + 255) / 256, 256, 0, stream>>>(W2, W2T, Hq, OUTq);

  // LTC recurrence
  ltc_kernel<<<Bq, Uq, 0, stream>>>(x, input_w, input_b, gleak, vleak, cm,
                                    sigma, mu, w, erev, mask,
                                    s_sigma, s_mu, s_w, s_erev, s_mask,
                                    output_w, output_b, ltc_f32, ltc_f16, h_final);

  // Q/K/V projections (WMMA + async LDS staging)
  dim3 gblk(32, 4);
  dim3 ggrid(Hq / 64, MQ / 64);
  gemm_kernel<false,false,false,true><<<ggrid, gblk, 0, stream>>>(ltc_f16, WqT, bq, nullptr, nullptr, Q16, Hq, Hq);
  gemm_kernel<false,false,false,true><<<ggrid, gblk, 0, stream>>>(ltc_f16, WkT, bk, nullptr, nullptr, K16, Hq, Hq);
  gemm_kernel<false,false,false,true><<<ggrid, gblk, 0, stream>>>(ltc_f16, WvT, bv, nullptr, nullptr, V16, Hq, Hq);

  // Hopfield retrieval
  attn_kernel<<<dim3(Bq, 4), dim3(32, 8), 0, stream>>>(Q16, K16, V16, mem16);

  // combined = ltc + mem@Wo + bo  (fp32 out + fp16 copy)
  gemm_kernel<true,false,true,true><<<ggrid, gblk, 0, stream>>>(mem16, WoT, bo, ltc_f32, combined, cmb16, Hq, Hq);
  // x1 = silu(combined@W1 + b1)
  gemm_kernel<false,true,false,true><<<ggrid, gblk, 0, stream>>>(cmb16, W1T, b1, nullptr, nullptr, x116, Hq, Hq);
  // belief = x1@W2 + b2
  gemm_kernel<false,false,true,false><<<dim3(OUTq / 64, MQ / 64), gblk, 0, stream>>>(x116, W2T, b2, nullptr, belief, nullptr, OUTq, Hq);
}
